// RandLANetUNet_68496138437076
// MI455X (gfx1250) — matrix operand order repaired and checked by reading, not compiled, not run on hardware
//
#include <hip/hip_runtime.h>
#include <math.h>

// ===========================================================================
// RandLA-Net U-Net forward for MI455X (gfx1250, wave32, WMMA).
//
// - All 1x1 convs / attention FCs run through one WMMA GEMM kernel:
//     Y[M,N] = act(X[M,C] @ W[C,N] + b), fp32 I/O, f16 multiply, f32 accum
//     (v_wmma_f32_16x16x32_f16). One wave = 16 rows x (NCH*16) cols.
//   K-loop is split: branch-free main loop over (C & ~31) with b128 loads,
//   plus a one-shot masked epilogue for the C%32 tail. N-edge handling is a
//   template parameter (every launch is either fully tiled or N<16).
// - Gathers / softmax-attention-pool / max-pool / interp are streaming
//   VALU kernels (bandwidth bound; HBM 23.3 TB/s is the ceiling there).
// - Batch (B=2) processed sequentially to halve (N,K,C) scratch.
// - Assumes params (d_in[21]) are the jax tree_flatten leaves (sorted dict
//   keys, lists in order, 'b' before 'w') concatenated as fp32.
// - Assumes ws_size >= ~246 MB (layout below).
// ===========================================================================

#define KNB 16

typedef __attribute__((ext_vector_type(16))) _Float16 v16h;
typedef __attribute__((ext_vector_type(8)))  float    v8f;

__device__ __forceinline__ void cvt16(v16h& d, const float4& p0, const float4& p1,
                                      const float4& p2, const float4& p3)
{
  d[0]  = (_Float16)p0.x; d[1]  = (_Float16)p0.y;
  d[2]  = (_Float16)p0.z; d[3]  = (_Float16)p0.w;
  d[4]  = (_Float16)p1.x; d[5]  = (_Float16)p1.y;
  d[6]  = (_Float16)p1.z; d[7]  = (_Float16)p1.w;
  d[8]  = (_Float16)p2.x; d[9]  = (_Float16)p2.y;
  d[10] = (_Float16)p2.z; d[11] = (_Float16)p2.w;
  d[12] = (_Float16)p3.x; d[13] = (_Float16)p3.y;
  d[14] = (_Float16)p3.z; d[15] = (_Float16)p3.w;
}

// ---------------------------------------------------------------------------
// WMMA GEMM. A-frag layout (16x32 f16, MxK): lane l holds row (l&15);
// element pair v holds K = (l&16?8:0) + 2*(v&3) + (v&4?16:0) (+0/+1).
// B-frag layout (32x16 f16, KxN): lane l holds N column (l&15); element s holds K = (l&16?8:0) + (s&7) + (s&8?16:0).
// D layout (16x16 f32): lane = N column (l&15), VGPR r = row r (+8 if l>=16).
// ---------------------------------------------------------------------------
template<int NCH, int ACT, bool NFULL>
__global__ __launch_bounds__(256) void gemm_wmma_f16(
    const float* __restrict__ X, const float* __restrict__ W,
    const float* __restrict__ Bv, float* __restrict__ Y,
    int M, int C, int N)
{
  const int lane = threadIdx.x & 31;
  const int wave = threadIdx.x >> 5;
  const int m0 = (blockIdx.x * 8 + wave) * 16;
  if (m0 >= M) return;
  const int n0 = blockIdx.y * (16 * NCH);

  v8f acc[NCH];
#pragma unroll
  for (int c = 0; c < NCH; ++c) acc[c] = (v8f){0.f,0.f,0.f,0.f,0.f,0.f,0.f,0.f};

  const int row = m0 + (lane & 15);
  const float* __restrict__ xr = X + (size_t)row * C;
  const int kb = (lane & 16) ? 8 : 0;
  const int kfull = C & ~31;

  // ---- branch-free main loop over full 32-wide K chunks ----
  for (int k0 = 0; k0 < kfull; k0 += 32) {
    __builtin_prefetch(xr + k0 + 32, 0, 3);   // global_prefetch_b8: next A chunk
    const float4 p0 = *(const float4*)(xr + k0 + kb);
    const float4 p1 = *(const float4*)(xr + k0 + kb + 4);
    const float4 p2 = *(const float4*)(xr + k0 + kb + 16);
    const float4 p3 = *(const float4*)(xr + k0 + kb + 20);
    v16h a;
    cvt16(a, p0, p1, p2, p3);

    const float* __restrict__ wr = W + (size_t)k0 * N;
#pragma unroll
    for (int c = 0; c < NCH; ++c) {
      v16h bfr;
      const int n = n0 + 16 * c + (lane & 15);
      if (NFULL) {
#pragma unroll
        for (int e = 0; e < 16; ++e) {
          const int kr = kb + (e & 7) + ((e & 8) << 1);
          bfr[e] = (_Float16)wr[(size_t)kr * N + n];
        }
      } else {
#pragma unroll
        for (int e = 0; e < 16; ++e) {
          const int kr = kb + (e & 7) + ((e & 8) << 1);
          const float f = wr[(size_t)kr * N + ((n < N) ? n : 0)];
          bfr[e] = (_Float16)((n < N) ? f : 0.0f);
        }
      }
      acc[c] = __builtin_amdgcn_wmma_f32_16x16x32_f16(
          false, a, false, bfr, (short)0, acc[c], false, false);
    }
  }

  // ---- one-shot masked epilogue for the K tail (C % 32 != 0) ----
  if (C & 31) {
    const int k0 = kfull;
    v16h a;
#pragma unroll
    for (int v = 0; v < 8; ++v) {
      const int k  = k0 + kb + ((v & 3) << 1) + ((v & 4) << 2);
      const int k1 = k + 1;
      const float f0 = xr[(k  < C) ? k  : 0];
      const float f1 = xr[(k1 < C) ? k1 : 0];
      a[2 * v]     = (_Float16)((k  < C) ? f0 : 0.0f);
      a[2 * v + 1] = (_Float16)((k1 < C) ? f1 : 0.0f);
    }
#pragma unroll
    for (int c = 0; c < NCH; ++c) {
      v16h bfr;
      const int n = n0 + 16 * c + (lane & 15);
#pragma unroll
      for (int e = 0; e < 16; ++e) {
        const int kk = k0 + kb + (e & 7) + ((e & 8) << 1);
        const bool kval = (kk < C);
        const float f = W[(size_t)(kval ? kk : 0) * N + ((n < N) ? n : 0)];
        bfr[e] = (_Float16)((kval && n < N) ? f : 0.0f);
      }
      acc[c] = __builtin_amdgcn_wmma_f32_16x16x32_f16(
          false, a, false, bfr, (short)0, acc[c], false, false);
    }
  }

  // ---- store D (+ bias, activation) ----
  const int nc = lane & 15;
  const int r0 = m0 + ((lane & 16) ? 8 : 0);
#pragma unroll
  for (int c = 0; c < NCH; ++c) {
    const int n = n0 + 16 * c + nc;
    if (!NFULL && n >= N) continue;
    const float bb = Bv ? Bv[n] : 0.0f;
#pragma unroll
    for (int r = 0; r < 8; ++r) {
      float y = acc[c][r] + bb;
      if (ACT == 1) y = (y > 0.0f) ? y : 0.0f;     // relu
      Y[(size_t)(r0 + r) * N + n] = y;
    }
  }
}

// ----------------------------- glue kernels --------------------------------

// fc0 + eval-BN: out = lrelu(f@W+b, 0.01) * gamma/sqrt(1+eps) + beta
__global__ void fc0_bn_kernel(const float* __restrict__ feats,
                              const float* __restrict__ w, const float* __restrict__ b,
                              const float* __restrict__ gamma, const float* __restrict__ beta,
                              float* __restrict__ out, int npts)
{
  int i = blockIdx.x * blockDim.x + threadIdx.x;
  if (i >= npts) return;
  float fin[6];
#pragma unroll
  for (int k = 0; k < 6; ++k) fin[k] = feats[(size_t)i * 6 + k];
  const float inv = 0.9999995000003750f;   // 1/sqrt(1 + 1e-6)
#pragma unroll
  for (int c = 0; c < 8; ++c) {
    float s = b[c];
#pragma unroll
    for (int k = 0; k < 6; ++k) s += fin[k] * w[k * 8 + c];
    s = (s > 0.0f) ? s : 0.01f * s;
    out[(size_t)i * 8 + c] = s * inv * gamma[c] + beta[c];
  }
}

// rel-pos encoding: out[(pt,k), 10] = [dist, rel(3), tile(3), nb(3)]
__global__ void relpos_kernel(const float* __restrict__ xyz, const int* __restrict__ nidx,
                              float* __restrict__ out, int n)
{
  int i = blockIdx.x * blockDim.x + threadIdx.x;   // pt*K + k
  if (i >= n * KNB) return;
  int pt = i >> 4;
  int j  = nidx[i];
  float tx = xyz[3 * pt], ty = xyz[3 * pt + 1], tz = xyz[3 * pt + 2];
  float nx = xyz[3 * j],  ny = xyz[3 * j + 1],  nz = xyz[3 * j + 2];
  float rx = tx - nx, ry = ty - ny, rz = tz - nz;
  float d = sqrtf(rx * rx + ry * ry + rz * rz);
  float* o = out + (size_t)i * 10;
  o[0] = d;  o[1] = rx; o[2] = ry; o[3] = rz;
  o[4] = tx; o[5] = ty; o[6] = tz; o[7] = nx; o[8] = ny; o[9] = nz;
}

// out[(pt,k), 0:cg] = feat[nidx[pt,k], :]; out[(pt,k), cg:cg+cx] = fx[(pt,k), :]
__global__ void gather_concat_kernel(const float* __restrict__ feat, int cg,
                                     const int* __restrict__ nidx,
                                     const float* __restrict__ fx, int cx,
                                     float* __restrict__ out, int n)
{
  const int ct = cg + cx;
  size_t total = (size_t)n * KNB * ct;
  size_t i = (size_t)blockIdx.x * blockDim.x + threadIdx.x;
  if (i >= total) return;
  int c = (int)(i % ct);
  size_t rk = i / ct;
  float v;
  if (c < cg) v = feat[(size_t)nidx[rk] * cg + c];
  else        v = fx[rk * cx + (c - cg)];
  out[i] = v;
}

// per-(pt,c) softmax over K and weighted sum: out[pt,c] = sum_k fset*softmax(logit)
__global__ void attn_pool_kernel(const float* __restrict__ fset,
                                 const float* __restrict__ logit,
                                 float* __restrict__ out, int n, int cdim)
{
  size_t total = (size_t)n * cdim;
  size_t i = (size_t)blockIdx.x * blockDim.x + threadIdx.x;
  if (i >= total) return;
  int c = (int)(i % cdim);
  size_t pt = i / cdim;
  size_t base = pt * (size_t)KNB * cdim + c;
  float mx = -3.0e38f;
#pragma unroll
  for (int k = 0; k < KNB; ++k) mx = fmaxf(mx, logit[base + (size_t)k * cdim]);
  float se = 0.0f, ac = 0.0f;
#pragma unroll
  for (int k = 0; k < KNB; ++k) {
    float e = __expf(logit[base + (size_t)k * cdim] - mx);
    se += e;
    ac += e * fset[base + (size_t)k * cdim];
  }
  out[i] = ac / se;
}

__global__ void residual_lrelu_kernel(const float* __restrict__ a, const float* __restrict__ b,
                                      float* __restrict__ out, size_t total)
{
  size_t i = (size_t)blockIdx.x * blockDim.x + threadIdx.x;
  if (i >= total) return;
  float v = a[i] + b[i];
  out[i] = (v > 0.0f) ? v : 0.2f * v;
}

// out[p,c] = max_k f[sidx[p,k], c]
__global__ void max_gather_kernel(const float* __restrict__ f, const int* __restrict__ sidx,
                                  float* __restrict__ out, int nd, int cdim)
{
  size_t total = (size_t)nd * cdim;
  size_t i = (size_t)blockIdx.x * blockDim.x + threadIdx.x;
  if (i >= total) return;
  int c = (int)(i % cdim);
  size_t p = i / cdim;
  const int* s = sidx + p * KNB;
  float m = -3.0e38f;
#pragma unroll
  for (int k = 0; k < KNB; ++k) m = fmaxf(m, f[(size_t)s[k] * cdim + c]);
  out[i] = m;
}

// out[p, 0:cs] = skip[p, :]; out[p, cs:cs+cu] = x[iidx[p], :]
__global__ void up_concat_kernel(const float* __restrict__ skip, int cs,
                                 const float* __restrict__ x, int cu,
                                 const int* __restrict__ iidx,
                                 float* __restrict__ out, int n)
{
  const int ct = cs + cu;
  size_t total = (size_t)n * ct;
  size_t i = (size_t)blockIdx.x * blockDim.x + threadIdx.x;
  if (i >= total) return;
  int c = (int)(i % ct);
  size_t p = i / ct;
  out[i] = (c < cs) ? skip[p * cs + c] : x[(size_t)iidx[p] * cu + (c - cs)];
}

// ----------------------------- host helpers --------------------------------

static inline unsigned eblocks(size_t n) { return (unsigned)((n + 255) / 256); }

static void gemm(const float* X, const float* W, const float* Bv, float* Y,
                 int M, int C, int N, int act, hipStream_t s)
{
  dim3 blk(256, 1, 1);
  unsigned gx = (unsigned)((M / 16 + 7) / 8);
  if (N >= 64) {
    dim3 g(gx, (unsigned)((N + 63) / 64));
    if (N % 64 == 0) {
      if (act) gemm_wmma_f16<4, 1, true><<<g, blk, 0, s>>>(X, W, Bv, Y, M, C, N);
      else     gemm_wmma_f16<4, 0, true><<<g, blk, 0, s>>>(X, W, Bv, Y, M, C, N);
    } else {
      if (act) gemm_wmma_f16<4, 1, false><<<g, blk, 0, s>>>(X, W, Bv, Y, M, C, N);
      else     gemm_wmma_f16<4, 0, false><<<g, blk, 0, s>>>(X, W, Bv, Y, M, C, N);
    }
  } else if (N >= 32) {
    dim3 g(gx, (unsigned)((N + 31) / 32));
    if (N % 32 == 0) {
      if (act) gemm_wmma_f16<2, 1, true><<<g, blk, 0, s>>>(X, W, Bv, Y, M, C, N);
      else     gemm_wmma_f16<2, 0, true><<<g, blk, 0, s>>>(X, W, Bv, Y, M, C, N);
    } else {
      if (act) gemm_wmma_f16<2, 1, false><<<g, blk, 0, s>>>(X, W, Bv, Y, M, C, N);
      else     gemm_wmma_f16<2, 0, false><<<g, blk, 0, s>>>(X, W, Bv, Y, M, C, N);
    }
  } else {
    dim3 g(gx, (unsigned)((N + 15) / 16));
    if (N % 16 == 0) {
      if (act) gemm_wmma_f16<1, 1, true><<<g, blk, 0, s>>>(X, W, Bv, Y, M, C, N);
      else     gemm_wmma_f16<1, 0, true><<<g, blk, 0, s>>>(X, W, Bv, Y, M, C, N);
    } else {
      if (act) gemm_wmma_f16<1, 1, false><<<g, blk, 0, s>>>(X, W, Bv, Y, M, C, N);
      else     gemm_wmma_f16<1, 0, false><<<g, blk, 0, s>>>(X, W, Bv, Y, M, C, N);
    }
  }
}

extern "C" void kernel_launch(void* const* d_in, const int* in_sizes, int n_in,
                              void* d_out, int out_size, void* d_ws, size_t ws_size,
                              hipStream_t stream)
{
  (void)in_sizes; (void)n_in; (void)out_size; (void)ws_size;

  const int D[5]    = {16, 64, 128, 256, 512};
  const int DIN[5]  = {8, 32, 128, 256, 512};
  const int NS[6]   = {65536, 16384, 4096, 1024, 512, 256};
  const int SKC[6]  = {32, 32, 128, 256, 512, 1024};

  const float* P = (const float*)d_in[21];

  // ---- parameter offsets: jax tree_flatten (sorted keys, 'b' < 'w') ----
  size_t off = 0;
  auto take = [&](size_t n) { size_t o = off; off += n; return o; };
  const int DEC_DI[5] = {1536, 768, 384, 160, 64};
  const int DEC_DO[5] = {512, 256, 128, 32, 32};
  size_t dec_b[5], dec_w[5];
  for (int j = 0; j < 5; ++j) { dec_b[j] = take(DEC_DO[j]); dec_w[j] = take((size_t)DEC_DI[j] * DEC_DO[j]); }
  size_t d0_b = take(1024), d0_w = take(1024 * 1024);
  size_t a1fc[5], a1m_b[5], a1m_w[5], a2fc[5], a2m_b[5], a2m_w[5];
  size_t b1_b[5], b1_w[5], b2_b[5], b2_w[5];
  size_t m1_b[5], m1_w[5], m2_b[5], m2_w[5], sc_b[5], sc_w[5];
  for (int i = 0; i < 5; ++i) {
    int d = D[i], di = DIN[i], h = d / 2;
    a1fc[i] = take((size_t)d * d);
    a1m_b[i] = take(h);      a1m_w[i] = take((size_t)d * h);
    a2fc[i] = take((size_t)d * d);
    a2m_b[i] = take(d);      a2m_w[i] = take((size_t)d * d);
    b1_b[i] = take(h);       b1_w[i] = take((size_t)10 * h);
    b2_b[i] = take(h);       b2_w[i] = take((size_t)h * h);
    m1_b[i] = take(h);       m1_w[i] = take((size_t)di * h);
    m2_b[i] = take(2 * d);   m2_w[i] = take((size_t)d * 2 * d);
    sc_b[i] = take(2 * d);   sc_w[i] = take((size_t)di * 2 * d);
  }
  size_t fc0_b = take(8), fc0_be = take(8), fc0_ga = take(8), fc0_w = take(48);
  size_t fc1_b = take(64), fc1_w = take(32 * 64);
  size_t fc2_b = take(32), fc2_w = take(64 * 32);
  size_t fc3_b = take(13), fc3_w = take(32 * 13);

  // ---- workspace layout (floats), ~246 MB; reused for both batches ----
  float* WSF = (float*)d_ws;
  size_t wo = 0;
  auto wtake = [&](size_t n) { float* p = WSF + wo; wo += n; return p; };
  float* SK[6];
  SK[0] = wtake((size_t)65536 * 32);
  SK[1] = wtake((size_t)16384 * 32);
  SK[2] = wtake((size_t)4096 * 128);
  SK[3] = wtake((size_t)1024 * 256);
  SK[4] = wtake((size_t)512 * 512);
  SK[5] = wtake((size_t)256 * 1024);
  float* X0   = wtake((size_t)65536 * 8);
  float* F1   = wtake(524288);      // max n*d/2
  float* POOL = wtake(1048576);     // max n*d
  float* FAGG = wtake(524288);
  float* FBB  = wtake(1048576);
  float* FML2 = wtake(2097152);     // max n*2d
  float* FSC  = wtake(2097152);
  float* FRES = wtake(2097152);
  float* BIG0 = wtake(16777216);    // max n*K*d
  float* BIG1 = wtake(16777216);
  float* BIG2 = wtake(16777216);
  float* DECA = BIG0;               // decoder phase reuses BIG regions
  float* DECB = BIG1;

  for (int b = 0; b < 2; ++b) {
    const float* feats = (const float*)d_in[0] + (size_t)b * 65536 * 6;
    const float* XYZ[5]; const int* NEI[5]; const int* SUBI[5]; const int* ITP[5];
    for (int i = 0; i < 5; ++i) {
      XYZ[i]  = (const float*)d_in[1 + 4 * i] + (size_t)b * NS[i] * 3;
      NEI[i]  = (const int*)d_in[2 + 4 * i] + (size_t)b * NS[i] * KNB;
      SUBI[i] = (const int*)d_in[3 + 4 * i] + (size_t)b * NS[i + 1] * KNB;
      ITP[i]  = (const int*)d_in[4 + 4 * i] + (size_t)b * NS[i];
    }
    float* outb = (float*)d_out + (size_t)b * 65536 * 13;

    // ---- fc0 + eval BN ----
    fc0_bn_kernel<<<eblocks(65536), 256, 0, stream>>>(
        feats, P + fc0_w, P + fc0_b, P + fc0_ga, P + fc0_be, X0, 65536);

    const float* x = X0;
    int din = 8;

    // ---- encoder ----
    for (int i = 0; i < 5; ++i) {
      const int d = D[i], h = d / 2, n = NS[i], nk = n * KNB;
      // mlp1
      gemm(x, P + m1_w[i], P + m1_b[i], F1, n, din, h, 1, stream);
      // relative position encoding -> BIG0 (nk,10)
      relpos_kernel<<<eblocks(nk), 256, 0, stream>>>(XYZ[i], NEI[i], BIG0, n);
      // bb_mlp1: (nk,10) -> (nk,h) in BIG1 (f_xyz)
      gemm(BIG0, P + b1_w[i], P + b1_b[i], BIG1, nk, 10, h, 1, stream);
      // f_cat = [gather(f1), f_xyz] -> BIG2 (nk,d)
      gather_concat_kernel<<<eblocks((size_t)nk * d), 256, 0, stream>>>(
          F1, h, NEI[i], BIG1, h, BIG2, n);
      // att1 logits -> BIG0 (rel-pos dead)
      gemm(BIG2, P + a1fc[i], nullptr, BIG0, nk, d, d, 0, stream);
      attn_pool_kernel<<<eblocks((size_t)n * d), 256, 0, stream>>>(BIG2, BIG0, POOL, n, d);
      gemm(POOL, P + a1m_w[i], P + a1m_b[i], FAGG, n, d, h, 1, stream);
      // bb_mlp2: f_xyz (BIG1) -> f_xyz2 (BIG2, f_cat dead)
      gemm(BIG1, P + b2_w[i], P + b2_b[i], BIG2, nk, h, h, 1, stream);
      // f_cat2 = [gather(f_agg), f_xyz2] -> BIG0 (logits dead)
      gather_concat_kernel<<<eblocks((size_t)nk * d), 256, 0, stream>>>(
          FAGG, h, NEI[i], BIG2, h, BIG0, n);
      // att2 logits -> BIG1 (f_xyz dead)
      gemm(BIG0, P + a2fc[i], nullptr, BIG1, nk, d, d, 0, stream);
      attn_pool_kernel<<<eblocks((size_t)n * d), 256, 0, stream>>>(BIG0, BIG1, POOL, n, d);
      gemm(POOL, P + a2m_w[i], P + a2m_b[i], FBB, n, d, d, 1, stream);
      // mlp2 (no act) + shortcut (no act) + residual leaky(0.2)
      gemm(FBB, P + m2_w[i], P + m2_b[i], FML2, n, d, 2 * d, 0, stream);
      gemm(x, P + sc_w[i], P + sc_b[i], FSC, n, din, 2 * d, 0, stream);
      float* fres = (i == 0) ? SK[0] : FRES;
      residual_lrelu_kernel<<<eblocks((size_t)n * 2 * d), 256, 0, stream>>>(
          FML2, FSC, fres, (size_t)n * 2 * d);
      // random-sample: gather + max over K -> skip[i+1]
      max_gather_kernel<<<eblocks((size_t)NS[i + 1] * 2 * d), 256, 0, stream>>>(
          fres, SUBI[i], SK[i + 1], NS[i + 1], 2 * d);
      x = SK[i + 1];
      din = 2 * d;
    }

    // ---- decoder ----
    gemm(SK[5], P + d0_w, P + d0_b, DECA, 256, 1024, 1024, 1, stream);
    int cu = 1024;
    for (int j = 0; j < 5; ++j) {
      const int lev = 4 - j;
      const int n = NS[lev];
      const int cs = SKC[lev];
      up_concat_kernel<<<eblocks((size_t)n * (cs + cu)), 256, 0, stream>>>(
          SK[lev], cs, DECA, cu, ITP[lev], DECB, n);
      gemm(DECB, P + dec_w[j], P + dec_b[j], DECA, n, cs + cu, DEC_DO[j], 1, stream);
      cu = DEC_DO[j];
    }
    // fc1 -> fc2 -> fc3 (logits)
    gemm(DECA, P + fc1_w, P + fc1_b, DECB, 65536, 32, 64, 1, stream);
    gemm(DECB, P + fc2_w, P + fc2_b, DECA, 65536, 64, 32, 1, stream);
    gemm(DECA, P + fc3_w, P + fc3_b, outb, 65536, 32, 13, 0, stream);
  }
}
